// SAGE_51823075393734
// MI455X (gfx1250) — compile-verified, matrix operand order checked by reading, not run on hardware
//
#include <hip/hip_runtime.h>

// GraphSAGE 2-layer forward for MI455X (gfx1250, wave32, WMMA).
//   Layer1: h = relu( mean_agg(x) @ w1_l^T + b1_l + x @ w1_r^T + b1_r )
//   Layer2: out =    ( mean_agg(h) @ w2_l^T + b2_l + h @ w2_r^T + b2_r )
//
// Structure:
//  - wave-per-edge gather + global_atomic_add_f32 into an L2-resident f32
//    accumulator (51 MB < 192 MB L2), degree computed once.
//  - one-shot f32->f16 conversion passes (mean 1/deg fused) so the GEMM
//    inner loops are pure 16B f16 loads feeding v_wmma_f32_16x16x32_f16.
//  - gemm1: one wave owns a 16x64 tile (A fragments reused by 4 N-subtiles).
//  - gemm2: one wave owns a 16x48 tile (full padded output row strip);
//    w2 padded to 48 rows of zeros so EXEC stays all-1s through WMMA.

typedef __attribute__((ext_vector_type(16))) _Float16 v16h;
typedef __attribute__((ext_vector_type(8)))  _Float16 v8h;
typedef __attribute__((ext_vector_type(4)))  _Float16 v4h;
typedef __attribute__((ext_vector_type(8)))  float    v8f;

#define SAGE_N_NODES 50000
#define SAGE_D       256    // in_feats == h_feats
#define SAGE_NCLS    40
#define SAGE_NPAD    48     // layer-2 N padded to 3 WMMA tiles

// ---------------------------------------------------------------------------
// f16 fragment loader (CDNA5 16-bit A/B layout for V_WMMA_F32_16X16X32_F16).
// Lane L: half = L>>4. Elems 0..7  <- K = k0 + 8*half + 0..7
//         Elems 8..15 <- K = k0 + 16 + 8*half + 0..7   (ISA 05_wmma §7.12.2)
// Both runs contiguous along K -> two 16B loads.
// ---------------------------------------------------------------------------
__device__ __forceinline__ v16h sage_frag_f16(const _Float16* __restrict__ row,
                                              int k0, int half) {
  v8h u0 = *(const v8h*)(row + k0 + 8 * half);
  v8h u1 = *(const v8h*)(row + k0 + 16 + 8 * half);
  v16h a;
#pragma unroll
  for (int j = 0; j < 8; ++j) { a[j] = u0[j]; a[8 + j] = u1[j]; }
  return a;
}

// ---------------------------------------------------------------------------
// Zero-fill scratch
// ---------------------------------------------------------------------------
__global__ void sage_zero_kernel(float* __restrict__ p, long long n) {
  long long i = (long long)blockIdx.x * blockDim.x + threadIdx.x;
  long long stride = (long long)gridDim.x * blockDim.x;
  for (; i < n; i += stride) p[i] = 0.0f;
}

// ---------------------------------------------------------------------------
// f32 -> f16 conversion, 4 elements/thread. If deg != nullptr, scales each
// element by 1/max(deg[row],1) with row = elementIndex / rowLen (mean fusion).
// ---------------------------------------------------------------------------
__global__ void sage_cvt_f16(const float* __restrict__ in,
                             _Float16* __restrict__ out,
                             long long n4, const float* __restrict__ deg,
                             int rowLen) {
  long long i = (long long)blockIdx.x * blockDim.x + threadIdx.x;
  long long stride = (long long)gridDim.x * blockDim.x;
  for (; i < n4; i += stride) {
    float4 v = ((const float4*)in)[i];
    float s = 1.0f;
    if (deg != nullptr) {
      long long row = (i * 4) / rowLen;
      s = 1.0f / fmaxf(deg[row], 1.0f);
    }
    v4h o;
    o[0] = (_Float16)(v.x * s);
    o[1] = (_Float16)(v.y * s);
    o[2] = (_Float16)(v.z * s);
    o[3] = (_Float16)(v.w * s);
    ((v4h*)out)[i] = o;
  }
}

// ---------------------------------------------------------------------------
// Edge scatter: wave per edge; each lane handles 8 consecutive features.
// f32 feature source (layer 1). Optionally accumulates degree.
// ---------------------------------------------------------------------------
__global__ void __launch_bounds__(256)
sage_scatter_f32(const float* __restrict__ feats,
                 const long long* __restrict__ src,
                 const long long* __restrict__ dst,
                 float* __restrict__ agg, float* __restrict__ deg,
                 int nEdges, int computeDeg) {
  int gtid = blockIdx.x * blockDim.x + threadIdx.x;
  int edge = gtid >> 5;            // wave-uniform
  int lane = gtid & 31;
  if (edge >= nEdges) return;
  long long s = src[edge];
  long long d = dst[edge];
  const float4* in = (const float4*)(feats + s * (long long)SAGE_D) + lane * 2;
  float4 a = in[0], b = in[1];
  float* outp = agg + d * (long long)SAGE_D + lane * 8;
  atomicAdd(outp + 0, a.x); atomicAdd(outp + 1, a.y);
  atomicAdd(outp + 2, a.z); atomicAdd(outp + 3, a.w);
  atomicAdd(outp + 4, b.x); atomicAdd(outp + 5, b.y);
  atomicAdd(outp + 6, b.z); atomicAdd(outp + 7, b.w);
  if (computeDeg && lane == 0) atomicAdd(deg + d, 1.0f);
}

// f16 feature source (layer 2, gathers h)
__global__ void __launch_bounds__(256)
sage_scatter_f16(const _Float16* __restrict__ feats,
                 const long long* __restrict__ src,
                 const long long* __restrict__ dst,
                 float* __restrict__ agg, int nEdges) {
  int gtid = blockIdx.x * blockDim.x + threadIdx.x;
  int edge = gtid >> 5;
  int lane = gtid & 31;
  if (edge >= nEdges) return;
  long long s = src[edge];
  long long d = dst[edge];
  v8h a = ((const v8h*)(feats + s * (long long)SAGE_D))[lane];  // 16B gather
  float* outp = agg + d * (long long)SAGE_D + lane * 8;
#pragma unroll
  for (int j = 0; j < 8; ++j) atomicAdd(outp + j, (float)a[j]);
}

// ---------------------------------------------------------------------------
// Layer-1 GEMM: h = relu( aggH @ wl^T + xH @ wr^T + b_l + b_r ), all-f16 in.
// One wave -> 16x64 tile (4 N-subtiles); A fragments reused 4x.
// ---------------------------------------------------------------------------
__global__ void __launch_bounds__(256)
sage_gemm1(const _Float16* __restrict__ aggH, const _Float16* __restrict__ xH,
           const _Float16* __restrict__ wl, const _Float16* __restrict__ wr,
           const float* __restrict__ b_l, const float* __restrict__ b_r,
           _Float16* __restrict__ h_out) {
  const int NG = SAGE_D / 64;                        // 4 groups of 64 cols
  int wave = (blockIdx.x * blockDim.x + threadIdx.x) >> 5;
  int lane = threadIdx.x & 31;
  int nWaves = (SAGE_N_NODES / 16) * NG;             // 12500
  if (wave >= nWaves) return;                        // wave-uniform exit
  int g = wave % NG;
  int tileM = wave / NG;
  int mn = lane & 15, half = lane >> 4;
  int m = tileM * 16 + mn;
  int n0 = g * 64 + mn;                              // lane's col in subtile 0

  const _Float16* aRow = aggH + (long long)m * SAGE_D;
  const _Float16* xRow = xH   + (long long)m * SAGE_D;
  const _Float16* wl0 = wl + (long long)n0 * SAGE_D; // B col n == W row n
  const _Float16* wr0 = wr + (long long)n0 * SAGE_D;

  v8f c[4] = {{}, {}, {}, {}};
  for (int k0 = 0; k0 < SAGE_D; k0 += 32) {
    v16h aA = sage_frag_f16(aRow, k0, half);
    v16h aX = sage_frag_f16(xRow, k0, half);
#pragma unroll
    for (int t = 0; t < 4; ++t) {
      v16h bL = sage_frag_f16(wl0 + (long long)t * 16 * SAGE_D, k0, half);
      c[t] = __builtin_amdgcn_wmma_f32_16x16x32_f16(false, aA, false, bL,
                                                    (short)0, c[t], false, false);
      v16h bR = sage_frag_f16(wr0 + (long long)t * 16 * SAGE_D, k0, half);
      c[t] = __builtin_amdgcn_wmma_f32_16x16x32_f16(false, aX, false, bR,
                                                    (short)0, c[t], false, false);
    }
  }

#pragma unroll
  for (int t = 0; t < 4; ++t) {
    int n = n0 + t * 16;
    float bias = b_l[n] + b_r[n];
#pragma unroll
    for (int r = 0; r < 8; ++r) {                    // C: VGPR r -> row r+8*half
      int mo = tileM * 16 + r + 8 * half;
      float v = fmaxf(c[t][r] + bias, 0.0f);         // ReLU
      h_out[(long long)mo * SAGE_D + n] = (_Float16)v;
    }
  }
}

// ---------------------------------------------------------------------------
// Layer-2 GEMM: out = aggH @ wl^T + h @ wr^T + b_l + b_r. One wave -> 16x48
// strip (N padded to 48 with zeroed weight rows); stores masked to n<40.
// ---------------------------------------------------------------------------
__global__ void __launch_bounds__(256)
sage_gemm2(const _Float16* __restrict__ aggH, const _Float16* __restrict__ h,
           const _Float16* __restrict__ wl, const _Float16* __restrict__ wr,
           const float* __restrict__ b_l, const float* __restrict__ b_r,
           float* __restrict__ out) {
  int wave = (blockIdx.x * blockDim.x + threadIdx.x) >> 5;
  int lane = threadIdx.x & 31;
  int nWaves = SAGE_N_NODES / 16;                    // 3125
  if (wave >= nWaves) return;
  int tileM = wave;
  int mn = lane & 15, half = lane >> 4;
  int m = tileM * 16 + mn;

  const _Float16* aRow = aggH + (long long)m * SAGE_D;
  const _Float16* hRow = h    + (long long)m * SAGE_D;
  const _Float16* wl0 = wl + (long long)mn * SAGE_D;  // padded to 48 rows
  const _Float16* wr0 = wr + (long long)mn * SAGE_D;

  v8f c[3] = {{}, {}, {}};
  for (int k0 = 0; k0 < SAGE_D; k0 += 32) {
    v16h aA = sage_frag_f16(aRow, k0, half);
    v16h aH = sage_frag_f16(hRow, k0, half);
#pragma unroll
    for (int t = 0; t < 3; ++t) {
      v16h bL = sage_frag_f16(wl0 + (long long)t * 16 * SAGE_D, k0, half);
      c[t] = __builtin_amdgcn_wmma_f32_16x16x32_f16(false, aA, false, bL,
                                                    (short)0, c[t], false, false);
      v16h bR = sage_frag_f16(wr0 + (long long)t * 16 * SAGE_D, k0, half);
      c[t] = __builtin_amdgcn_wmma_f32_16x16x32_f16(false, aH, false, bR,
                                                    (short)0, c[t], false, false);
    }
  }

#pragma unroll
  for (int t = 0; t < 3; ++t) {
    int n = t * 16 + mn;
    if (n < SAGE_NCLS) {                             // mask only the store
      float bias = b_l[n] + b_r[n];
#pragma unroll
      for (int r = 0; r < 8; ++r) {
        int mo = tileM * 16 + r + 8 * half;
        out[(long long)mo * SAGE_NCLS + n] = c[t][r] + bias;
      }
    }
  }
}

// ---------------------------------------------------------------------------
extern "C" void kernel_launch(void* const* d_in, const int* in_sizes, int n_in,
                              void* d_out, int out_size, void* d_ws, size_t ws_size,
                              hipStream_t stream) {
  const float*     x    = (const float*)d_in[0];
  const long long* ei   = (const long long*)d_in[1];   // int64 per reference
  const float*     w1_l = (const float*)d_in[2];
  const float*     b1_l = (const float*)d_in[3];
  const float*     w1_r = (const float*)d_in[4];
  const float*     b1_r = (const float*)d_in[5];
  const float*     w2_l = (const float*)d_in[6];
  const float*     b2_l = (const float*)d_in[7];
  const float*     w2_r = (const float*)d_in[8];
  const float*     b2_r = (const float*)d_in[9];

  int nEdges = in_sizes[1] / 2;                 // edge_index is [2, E] flat
  const long long* src = ei;
  const long long* dst = ei + nEdges;

  // Workspace layout (all sections 256B-aligned):
  char* ws = (char*)d_ws;
  size_t off = 0;
  float* agg = (float*)(ws + off);              // f32 [N,256] sum accumulator
  off += (size_t)SAGE_N_NODES * SAGE_D * sizeof(float);            // 51.2 MB
  float* deg = (float*)(ws + off);              // f32 [N]
  off += ((size_t)SAGE_N_NODES * sizeof(float) + 255) & ~(size_t)255;
  _Float16* hb = (_Float16*)(ws + off);         // f16 [N,256] layer-1 output
  off += (size_t)SAGE_N_NODES * SAGE_D * sizeof(_Float16);         // 25.6 MB
  _Float16* aggH = (_Float16*)(ws + off);       // f16 [N,256] mean-scaled agg
  off += (size_t)SAGE_N_NODES * SAGE_D * sizeof(_Float16);         // 25.6 MB
  _Float16* xH = (_Float16*)(ws + off);         // f16 [N,256] x
  off += (size_t)SAGE_N_NODES * SAGE_D * sizeof(_Float16);         // 25.6 MB
  _Float16* w1lH = (_Float16*)(ws + off);       // f16 [256,256]
  off += (size_t)SAGE_D * SAGE_D * sizeof(_Float16);
  _Float16* w1rH = (_Float16*)(ws + off);       // f16 [256,256]
  off += (size_t)SAGE_D * SAGE_D * sizeof(_Float16);
  _Float16* w2lH = (_Float16*)(ws + off);       // f16 [48,256] zero-padded
  off += (size_t)SAGE_NPAD * SAGE_D * sizeof(_Float16);
  _Float16* w2rH = (_Float16*)(ws + off);       // f16 [48,256] zero-padded
  off += (size_t)SAGE_NPAD * SAGE_D * sizeof(_Float16);

  const int THR = 256;
  long long aggN = (long long)SAGE_N_NODES * SAGE_D;
  long long zeroN1 = aggN + (50176);             // agg + padded deg region
  int scatterBlocks = (nEdges * 32 + THR - 1) / THR;
  int cvtBigBlocks  = 2048;
  int gemm1Blocks = ((SAGE_N_NODES / 16) * 4 + 7) / 8;   // 12500 waves
  int gemm2Blocks = ((SAGE_N_NODES / 16) + 7) / 8;       //  3125 waves

  // ---- One-time f16 conversions (weights, x) ----
  sage_cvt_f16<<<64, THR, 0, stream>>>(w1_l, w1lH,
      (long long)SAGE_D * SAGE_D / 4, nullptr, SAGE_D);
  sage_cvt_f16<<<64, THR, 0, stream>>>(w1_r, w1rH,
      (long long)SAGE_D * SAGE_D / 4, nullptr, SAGE_D);
  // zero padded w2 buffers (2 x 48x256 f16 = 12288 floats), then convert 40 rows
  sage_zero_kernel<<<32, THR, 0, stream>>>((float*)w2lH,
      (long long)2 * SAGE_NPAD * SAGE_D * 2 / 4);
  sage_cvt_f16<<<16, THR, 0, stream>>>(w2_l, w2lH,
      (long long)SAGE_NCLS * SAGE_D / 4, nullptr, SAGE_D);
  sage_cvt_f16<<<16, THR, 0, stream>>>(w2_r, w2rH,
      (long long)SAGE_NCLS * SAGE_D / 4, nullptr, SAGE_D);
  sage_cvt_f16<<<cvtBigBlocks, THR, 0, stream>>>(x, xH, aggN / 4, nullptr, SAGE_D);

  // ---- Layer 1 ----
  sage_zero_kernel<<<2048, THR, 0, stream>>>(agg, zeroN1);           // agg + deg
  sage_scatter_f32<<<scatterBlocks, THR, 0, stream>>>(x, src, dst, agg, deg,
                                                      nEdges, 1);
  sage_cvt_f16<<<cvtBigBlocks, THR, 0, stream>>>(agg, aggH, aggN / 4,
                                                 deg, SAGE_D);       // mean fused
  sage_gemm1<<<gemm1Blocks, THR, 0, stream>>>(aggH, xH, w1lH, w1rH,
                                              b1_l, b1_r, hb);

  // ---- Layer 2 ----
  sage_zero_kernel<<<2048, THR, 0, stream>>>(agg, aggN);             // agg only
  sage_scatter_f16<<<scatterBlocks, THR, 0, stream>>>(hb, src, dst, agg, nEdges);
  sage_cvt_f16<<<cvtBigBlocks, THR, 0, stream>>>(agg, aggH, aggN / 4,
                                                 deg, SAGE_D);       // mean fused
  sage_gemm2<<<gemm2Blocks, THR, 0, stream>>>(aggH, hb, w2lH, w2rH,
                                              b2_l, b2_r, (float*)d_out);
}
